// WeightSupConLoss_22299470201110
// MI455X (gfx1250) — matrix operand order repaired and checked by reading, not compiled, not run on hardware
//
#include <hip/hip_runtime.h>
#include <hip/hip_bf16.h>
#include <math.h>

typedef __attribute__((ext_vector_type(16))) __bf16   v16bf;
typedef __attribute__((ext_vector_type(8)))  float    v8f;
typedef __attribute__((ext_vector_type(4)))  unsigned u32x4;
typedef __attribute__((ext_vector_type(8)))  int      i32x8;
typedef __attribute__((ext_vector_type(4)))  int      i32x4;

#define BSZ    4096
#define NTOT   8192              // n_views * bsz
#define DIM    128
#define NSTRIP 4                 // column strips per row-block
#define TPS    (NTOT / 16 / NSTRIP)   // tiles per strip = 128
#define INV_T  14.2857142857142857f   // 1 / 0.07

// ---------------------------------------------------------------------------
// Kernel 1: reindex features [bsz, n_views, dim] -> Z[v*bsz + b][k] as bf16,
//           and zero the scalar accumulator.
// ---------------------------------------------------------------------------
__global__ void supcon_convert(const float* __restrict__ feat,
                               __hip_bfloat16* __restrict__ Z,
                               float* __restrict__ acc)
{
    int t = blockIdx.x * blockDim.x + threadIdx.x;   // 0 .. NTOT*DIM-1
    if (t == 0) acc[0] = 0.0f;
    int r = t >> 7;           // row in Z (0..8191)
    int k = t & (DIM - 1);    // dim index
    int b = r & (BSZ - 1);    // batch index
    int v = r >> 12;          // view index (0/1)
    float x = feat[((b << 1) + v) * DIM + k];
    Z[r * DIM + k] = __float2bfloat16(x);
}

// ---------------------------------------------------------------------------
// TDM: stage one 16-row x 128-col bf16 tile (4 KB) of Z into LDS.
// D# group0: count=1, lds_addr, 57-bit global addr, type=2.
// D# group1: data_size=2B, tensor_dim0=128, tensor_dim1=8192,
//            tile_dim0=128, tile_dim1=16, tensor_dim0_stride=128.
// Groups 2/3 zero (tile_dim2=0 -> dims unused); trailing group zero.
// ---------------------------------------------------------------------------
__device__ __forceinline__ void tdm_load_tile(unsigned lds_off,
                                              const __hip_bfloat16* gsrc)
{
    unsigned long long ga = (unsigned long long)(uintptr_t)gsrc;
    u32x4 g0 = { 1u,                                   // count = 1 descriptor
                 lds_off,                              // LDS byte address
                 (unsigned)(ga & 0xFFFFFFFFull),
                 (unsigned)((ga >> 32) & 0x1FFFFFFull) | (2u << 30) };  // type=2
    i32x8 g1 = { (int)(1u << 16),        // data_size = 2 bytes
                 (int)(128u << 16),      // tensor_dim0 = 128 (bits 79:48 lo16)
                 (int)(0x2000u << 16),   // tensor_dim1 = 8192 (bits 111:80 lo16)
                 (int)(128u << 16),      // tile_dim0 = 128 (bits 127:112)
                 16,                     // tile_dim1 = 16   (bits 143:128)
                 128,                    // tensor_dim0_stride = 128
                 0, 0 };
    i32x4 gz4 = { 0, 0, 0, 0 };
    i32x8 gz8 = { 0, 0, 0, 0, 0, 0, 0, 0 };
    __builtin_amdgcn_tensor_load_to_lds(g0, g1, gz4, gz4, gz8, 0);
}

// ---------------------------------------------------------------------------
// Kernel 2: block = 128 anchor rows x 1 column strip (128 tiles).
// Wave w owns rows rb*128 + w*16 .. +15. Column tiles are TDM-staged into a
// double-buffered LDS tile shared by all 8 waves.
//   A (M x K) = contrast-column tile (from LDS)
//   B (K x N) = anchor-row tile (per wave, from global, loaded once)
//   D: lane l -> N = l&15 (row), VGPR r -> M = r + 8*(l>>4) (col)
// Per-row online-softmax partials go to ws; strips merged by supcon_merge.
// ---------------------------------------------------------------------------
__global__ __launch_bounds__(256) void supcon_main(
    const __hip_bfloat16* __restrict__ Zh,
    const float* __restrict__ W,
    const int*  __restrict__ labels,
    float4* __restrict__ part)
{
    const int strip = blockIdx.x & (NSTRIP - 1);
    const int rb    = blockIdx.x >> 2;          // row block (0..63)
    const int lane  = threadIdx.x & 31;
    const int wave  = threadIdx.x >> 5;         // 0..7
    const int nlow  = lane & 15;
    const int hi    = lane >> 4;

    __shared__ __align__(16) unsigned tilebuf[2][1024];   // 2 x 4 KB

    const unsigned* Zu = (const unsigned*)Zh;   // 2 bf16 per dword
    const int rowpitch = DIM / 2;               // 64 dwords per Z row

    union Frag { v16bf v; unsigned u[8]; };

    // --- B fragments: this wave's anchor-row tile (fixed, from global) -----
    // 16-bit B layout (K x 16): lane n = l&15 is N; lanes 0-15 hold K=0..7,16..23
    // of each 32-chunk, lanes 16-31 K=8..15,24..31.
    Frag bfr[4];
    const int myrow = rb * 128 + wave * 16 + nlow;   // this lane's anchor row
    {
#pragma unroll
        for (int kc = 0; kc < 4; ++kc)
#pragma unroll
            for (int d = 0; d < 8; ++d) {
                int k = kc * 32 + (d >> 2) * 16 + hi * 8 + (d & 3) * 2;
                bfr[kc].u[d] = Zu[myrow * rowpitch + (k >> 1)];
            }
    }

    const int labr     = labels[myrow & (BSZ - 1)];
    const size_t wbase = (size_t)(myrow & (BSZ - 1)) * BSZ;

    float mx = -INFINITY, sm = 0.0f, pa = 0.0f, pm = 0.0f;

    const int ct0 = strip * TPS;
    const unsigned lds0 = (unsigned)(uintptr_t)&tilebuf[0][0];
    const unsigned lds1 = (unsigned)(uintptr_t)&tilebuf[1][0];

    // preload tile 0 of this strip
    if (wave == 0)
        tdm_load_tile(lds0, Zh + (size_t)ct0 * 16 * DIM);

    for (int i = 0; i < TPS; ++i) {
        const int ct  = ct0 + i;
        const int cur = i & 1;

        if (wave == 0) __builtin_amdgcn_s_wait_tensorcnt(0);  // tile i arrived
        __syncthreads();   // tile i visible; buffers from iter i-1 free

        if (wave == 0 && i + 1 < TPS)
            tdm_load_tile((i & 1) ? lds0 : lds1,
                          Zh + (size_t)(ct + 1) * 16 * DIM);

        // --- A fragments from LDS (16-bit A layout) ------------------------
        // lane m = l&15 is M; lanes 0-15: K=0..7 (V0-3), 16..23 (V4-7);
        // lanes 16-31: K=8..15, 24..31. Dwords are two contiguous b128 runs.
        const unsigned* Ls = &tilebuf[cur][0];
        Frag afr[4];
#pragma unroll
        for (int kc = 0; kc < 4; ++kc) {
            const uint4 lo4 = *(const uint4*)(Ls + nlow * 64 + kc * 16 + hi * 4);
            const uint4 hi4 = *(const uint4*)(Ls + nlow * 64 + kc * 16 + 8 + hi * 4);
            afr[kc].u[0] = lo4.x; afr[kc].u[1] = lo4.y;
            afr[kc].u[2] = lo4.z; afr[kc].u[3] = lo4.w;
            afr[kc].u[4] = hi4.x; afr[kc].u[5] = hi4.y;
            afr[kc].u[6] = hi4.z; afr[kc].u[7] = hi4.w;
        }

        v8f c = {};
#pragma unroll
        for (int kc = 0; kc < 4; ++kc)
            c = __builtin_amdgcn_wmma_f32_16x16x32_bf16(
                    false, afr[kc].v, false, bfr[kc].v,
                    (short)0, c, false, false);

        // --- epilogue: this lane = (row myrow, cols ct*16+8*hi + 0..7) -----
        const int colbase = ct * 16 + 8 * hi;
        const int cb      = colbase & (BSZ - 1);     // 32B-aligned
        const float4 w0 = *(const float4*)(W + wbase + cb);
        const float4 w1 = *(const float4*)(W + wbase + cb + 4);
        const int4   l0 = *(const int4*)(labels + cb);
        const int4   l1 = *(const int4*)(labels + cb + 4);
        const float wv[8] = { w0.x, w0.y, w0.z, w0.w, w1.x, w1.y, w1.z, w1.w };
        const int   lv[8] = { l0.x, l0.y, l0.z, l0.w, l1.x, l1.y, l1.z, l1.w };

        float dv[8];
#pragma unroll
        for (int r = 0; r < 8; ++r) dv[r] = c[r] * INV_T;

        float t01 = fmaxf(dv[0], dv[1]), t23 = fmaxf(dv[2], dv[3]);
        float t45 = fmaxf(dv[4], dv[5]), t67 = fmaxf(dv[6], dv[7]);
        float tmax = fmaxf(fmaxf(t01, t23), fmaxf(t45, t67));
        float nm   = fmaxf(mx, tmax);

        float ssum = 0.0f;
#pragma unroll
        for (int r = 0; r < 8; ++r) {
            const bool diag = (colbase + r == myrow);
            ssum += diag ? 0.0f : __expf(dv[r] - nm);
            float msk = (lv[r] == labr && !diag) ? wv[r] : 0.0f;
            pa += msk * dv[r];
            pm += msk;
        }
        sm = sm * __expf(mx - nm) + ssum;   // one rescale per 8 elements
        mx = nm;
    }

    // --- merge the two half-wave column subsets of this row ----------------
    {
        float m2 = __shfl_xor(mx, 16, 32);
        float s2 = __shfl_xor(sm, 16, 32);
        float nm = fmaxf(mx, m2);
        sm = sm * __expf(mx - nm) + s2 * __expf(m2 - nm);
        mx = nm;
        pa += __shfl_xor(pa, 16, 32);
        pm += __shfl_xor(pm, 16, 32);
    }

    // each wave owns distinct rows -> write per-row strip partial directly
    if (lane < 16)
        part[(size_t)strip * NTOT + myrow] = make_float4(mx, sm, pa, pm);
}

// ---------------------------------------------------------------------------
// Kernel 3: merge the NSTRIP partials per row (online log-sum-exp), compute
//           mean_log_prob_pos and accumulate.
// ---------------------------------------------------------------------------
__global__ void supcon_merge(const float4* __restrict__ part,
                             float* __restrict__ acc)
{
    const int row = blockIdx.x * blockDim.x + threadIdx.x;   // 0..NTOT-1
    float4 p = part[row];
    float M = p.x, S = p.y, A = p.z, P = p.w;
#pragma unroll
    for (int s = 1; s < NSTRIP; ++s) {
        float4 q = part[(size_t)s * NTOT + row];
        float nm = fmaxf(M, q.x);
        S = S * __expf(M - nm) + q.y * __expf(q.x - nm);
        M = nm;
        A += q.z;
        P += q.w;
    }
    float denom = (P < 1e-6f) ? 1.0f : P;
    float mlpp  = (A - P * (M + __logf(S))) / denom;
    atomicAdd(acc, mlpp);
}

// ---------------------------------------------------------------------------
// Kernel 4: loss = -(T/BT) * mean(mlpp) / 2 = -acc / (2 * NTOT)   (T/BT == 1)
// ---------------------------------------------------------------------------
__global__ void supcon_finalize(const float* __restrict__ acc,
                                float* __restrict__ out)
{
    if (threadIdx.x == 0 && blockIdx.x == 0)
        out[0] = -acc[0] * (1.0f / (2.0f * (float)NTOT));
}

extern "C" void kernel_launch(void* const* d_in, const int* in_sizes, int n_in,
                              void* d_out, int out_size, void* d_ws, size_t ws_size,
                              hipStream_t stream)
{
    (void)in_sizes; (void)n_in; (void)out_size; (void)ws_size;

    const float* feat   = (const float*)d_in[0];   // [4096, 2, 128] f32
    const float* weight = (const float*)d_in[1];   // [4096, 4096]  f32
    const int*   labels = (const int*)  d_in[2];   // [4096] int32

    float* out = (float*)d_out;
    float* acc = (float*)d_ws;                                   // 1 float
    __hip_bfloat16* Z = (__hip_bfloat16*)((char*)d_ws + 256);    // 2 MB bf16
    float4* part = (float4*)((char*)d_ws + 256 + NTOT * DIM * 2); // 512 KB

    // 1) reindex + fp32->bf16 convert, zero accumulator
    supcon_convert<<<(NTOT * DIM) / 256, 256, 0, stream>>>(feat, Z, acc);

    // 2) fused WMMA Gram matrix + online softmax, TDM-staged column tiles
    supcon_main<<<(NTOT / 128) * NSTRIP, 256, 0, stream>>>(Z, weight, labels, part);

    // 3) merge strip partials per row
    supcon_merge<<<NTOT / 256, 256, 0, stream>>>(part, acc);

    // 4) scale to final scalar loss
    supcon_finalize<<<1, 1, 0, stream>>>(acc, out);
}
